// CharRNN_56513179681038
// MI455X (gfx1250) — matrix operand order repaired and checked by reading, not compiled, not run on hardware
//
#include <hip/hip_runtime.h>

typedef __attribute__((ext_vector_type(16))) __bf16 v16bf;
typedef __attribute__((ext_vector_type(8)))  float  v8f;

#define TLEN 512
#define EMBD 64
#define CHID 128
#define HIDN 256
#define NG   768
#define NCH  128
#define NCHP 129   // 128 chars + zero sentinel row (branchless SAME padding)
#define CPAD 516   // char window: 2 lead + 512 + 2 tail sentinels

struct f3 { float x, y, z; };   // packed (z,r,h) triple for one hidden column

// ---- gfx1250 native tanh if available; exp2+rcp fallback (no IEEE div) ----
__device__ __forceinline__ float fast_tanh(float x) {
#if __has_builtin(__builtin_amdgcn_tanhf)
  return __builtin_amdgcn_tanhf(x);
#else
  float e = __builtin_amdgcn_exp2f(x * 2.8853900817779268f);
  return 1.0f - 2.0f * __builtin_amdgcn_rcpf(e + 1.0f);
#endif
}
__device__ __forceinline__ float fast_sigmoid(float x) {
  return fmaf(0.5f, fast_tanh(0.5f * x), 0.5f);
}

// A-operand slab loader: two 16B LDS chunks (matches ds_load_b128 x2 codegen)
__device__ __forceinline__ v16bf load_a_slab(const __bf16* hbrow, int kc, int hi) {
  union { uint4 q[2]; v16bf v; } u;
  const char* base = (const char*)hbrow + (kc << 6) + (hi << 5);
  u.q[0] = *(const uint4*)(base);
  u.q[1] = *(const uint4*)(base + 16);
  return u.v;
}

// B-operand slab loader from global (row = Whb[n] slice for this lane)
__device__ __forceinline__ v16bf load_b_slab(const __bf16* row, int kc) {
  union { uint4 q[2]; v16bf v; } u;
  const uint4* p = (const uint4*)(row + (kc << 5));
  u.q[0] = p[0];
  u.q[1] = p[1];
  return u.v;
}

// ---------------------------------------------------------------------------
// GRU recurrence: 8 blocks (16 batch rows each) x 512 threads (16 waves).
// Wave w owns hidden columns j in [16w,16w+16) and computes all three gate
// columns (j, 256+j, 512+j) so the nonlinear update is wave-local.
// Gates r/hh Wh^T slices are register-resident (128 VGPRs); gate z's slice
// is streamed from the L2-hot Whb each step. The input-projection gather
// reads one packed b96 triple per (row, offset): 40 loads/step, coalesced.
// ---------------------------------------------------------------------------
__global__ __launch_bounds__(512, 1)
void k_gru(const int*   __restrict__ X,      // [128][512]
           const float* __restrict__ P,      // [5][129][256][3] packed projections
           const float* __restrict__ biasx,  // [768] = b_in + conv_bias@Wx
           const __bf16* __restrict__ Whb,   // [768][256] = Wh^T in bf16
           const float* __restrict__ brec,   // [768]
           float*       __restrict__ out)    // [128][256]
{
  __shared__ int    chars[16][CPAD];      // ~33 KB, sentinel-padded
  __shared__ __bf16 hb[16][HIDN + 8];     // ~8.3 KB: bf16 h (WMMA A operand)

  const int b0   = blockIdx.x * 16;
  const int tid  = threadIdx.x;
  const int wave = tid >> 5;
  const int lane = tid & 31;
  const int lo   = lane & 15;
  const int hi   = lane >> 4;
  const int j    = (wave << 4) + lo;      // hidden column owned by this lane

  for (int idx = tid; idx < 16 * CPAD; idx += 512) {
    int m = idx / CPAD, tt = idx % CPAD;
    chars[m][tt] = (tt >= 2 && tt < TLEN + 2)
                     ? X[(size_t)(b0 + m) * TLEN + (tt - 2)]
                     : NCH;               // sentinel -> zero row of P
  }
  for (int idx = tid; idx < 16 * HIDN; idx += 512) {
    hb[idx >> 8][idx & (HIDN - 1)] = (__bf16)0.0f;
  }
  __syncthreads();

  // ---- Register-resident Wh^T slices for gates r (1) and hh (2).
  v16bf breg1[8], breg2[8];
#pragma unroll
  for (int kc = 0; kc < 8; ++kc) {
    breg1[kc] = load_b_slab(Whb + (size_t)(1 * HIDN + j) * HIDN + (hi << 4), kc);
    breg2[kc] = load_b_slab(Whb + (size_t)(2 * HIDN + j) * HIDN + (hi << 4), kc);
  }
  // Gate z's slice row base: streamed per step (addresses invariant, L2-hot).
  const __bf16* WzRow = Whb + (size_t)j * HIDN + (hi << 4);

  // b_rec folded into the z/r gather bias; hh keeps b_rec separate from xh.
  const float bxz = biasx[j] + brec[j];
  const float bxr = biasx[HIDN + j] + brec[HIDN + j];
  const float bxh = biasx[2 * HIDN + j];
  const float brh = brec[2 * HIDN + j];
  // Column-packed P base for this lane; per-offset displacement is a
  // compile-time constant that folds into the 24-bit instruction offset.
  const float* Pj = P + (size_t)j * 3;

  float hreg[8];                // fp32 master state for (m = i+8*hi, j)
#pragma unroll
  for (int i = 0; i < 8; ++i) hreg[i] = 0.0f;

  for (int t = 0; t < TLEN; ++t) {
    // ---- Branchless gather: one packed b96 triple per (row, offset).
    v8f acc0, acc1, acc2;
    float xh[8];
#pragma unroll
    for (int i = 0; i < 8; ++i) {
      const int m = i + (hi << 3);
      float sz = bxz, sr = bxr, sh = bxh;
#pragma unroll
      for (int o = 0; o < 5; ++o) {
        const float* rowp = Pj + (size_t)chars[m][t + o] * (HIDN * 3);
        const f3 v = *(const f3*)(rowp + (size_t)o * NCHP * HIDN * 3);
        sz += v.x; sr += v.y; sh += v.z;
      }
      acc0[i] = sz;             // xz + b_in' + b_rec   (GEMM adds hz)
      acc1[i] = sr;             // xr + ...             (GEMM adds hr)
      acc2[i] = brh;            // hh = h@Wh + b_rec    (xh kept separate)
      xh[i]   = sh;
    }

    // ---- hw = h @ Wh : 8 k-slabs of 32, double-buffered A and streamed Bz.
    //      Per slab: acc2/acc1 use resident B; streamed-B acc0 goes last.
    {
      const __bf16* hbrow = &hb[lo][0];
      v16bf a[2], bz[2];
      a[0]  = load_a_slab(hbrow, 0, hi);
      bz[0] = load_b_slab(WzRow, 0);
#pragma unroll
      for (int kc = 0; kc < 8; ++kc) {
        if (kc + 1 < 8) {
          a[(kc + 1) & 1]  = load_a_slab(hbrow, kc + 1, hi);
          bz[(kc + 1) & 1] = load_b_slab(WzRow, kc + 1);
        }
        const v16bf av = a[kc & 1];
        acc2 = __builtin_amdgcn_wmma_f32_16x16x32_bf16(false, av, false, breg2[kc],  (short)0, acc2, false, false);
        acc1 = __builtin_amdgcn_wmma_f32_16x16x32_bf16(false, av, false, breg1[kc],  (short)0, acc1, false, false);
        acc0 = __builtin_amdgcn_wmma_f32_16x16x32_bf16(false, av, false, bz[kc & 1], (short)0, acc0, false, false);
      }
    }

    // ---- Wave-local gate fusion + state update (native v_tanh path).
    float hnew[8];
#pragma unroll
    for (int i = 0; i < 8; ++i) {
      const float z  = fast_sigmoid(acc0[i]);
      const float r  = fast_sigmoid(acc1[i]);
      const float hc = fast_tanh(fmaf(r, acc2[i], xh[i]));
      hnew[i] = z * hreg[i] + (1.0f - z) * hc;
      hreg[i] = hnew[i];                  // fp32 master state in registers
    }
    __syncthreads();                      // all waves done reading hb this step
#pragma unroll
    for (int i = 0; i < 8; ++i) hb[i + (hi << 3)][j] = (__bf16)hnew[i];
    __syncthreads();                      // hb ready for next step's A loads
  }

#pragma unroll
  for (int i = 0; i < 8; ++i) {
    const int m = i + (hi << 3);
    out[(size_t)(b0 + m) * HIDN + j] = hreg[i];
  }
}

// ---------------------------------------------------------------------------
// Precompute E_o[e][n] = sum over (k,tap) at offset o of conv_w_k[tap] @ Wx_blk
// ---------------------------------------------------------------------------
__global__ void k_eff(const float* __restrict__ w2, const float* __restrict__ w3,
                      const float* __restrict__ w4, const float* __restrict__ w5,
                      const float* __restrict__ Wx, float* __restrict__ E)
{
  int idx = blockIdx.x * blockDim.x + threadIdx.x;
  if (idx >= 5 * EMBD * NG) return;
  int n = idx % NG;
  int e = (idx / NG) % EMBD;
  int o = idx / (NG * EMBD);                 // 0..4 -> offset o-2
  const float* wk[4]   = {w2, w3, w4, w5};
  const int    klen[4] = {2, 3, 4, 5};
  const int    padl[4] = {0, 1, 1, 2};       // SAME: pad_lo = (k-1)//2
  float acc = 0.0f;
  for (int kk = 0; kk < 4; ++kk) {
    int tap = (o - 2) + padl[kk];
    if (tap < 0 || tap >= klen[kk]) continue;
    const float* w  = wk[kk] + ((size_t)tap * EMBD + e) * CHID;  // w[tap][e][c]
    const float* wx = Wx + (size_t)(kk * CHID) * NG + n;         // Wx[kk*128+c][n]
    float s = 0.0f;
    for (int c = 0; c < CHID; ++c) s += w[c] * wx[(size_t)c * NG];
    acc += s;
  }
  E[idx] = acc;
}

// Packed P: P[((o*129 + c)*256 + n0)*3 + g] = emb[c] @ E_o[:, g*256+n0]
// Row c==128 is the zero sentinel.
__global__ void k_ptab(const float* __restrict__ emb, const float* __restrict__ E,
                       float* __restrict__ P)
{
  int idx = blockIdx.x * blockDim.x + threadIdx.x;
  if (idx >= 5 * NCHP * NG) return;
  int g  = idx % 3;
  int n0 = (idx / 3) % HIDN;
  int c  = (idx / (3 * HIDN)) % NCHP;
  int o  = idx / (3 * HIDN * NCHP);
  if (c == NCH) { P[idx] = 0.0f; return; }
  const int n = g * HIDN + n0;               // original column in [0,768)
  const float* Eo = E + (size_t)o * EMBD * NG;
  float s = 0.0f;
  for (int e = 0; e < EMBD; ++e) s += emb[c * EMBD + e] * Eo[(size_t)e * NG + n];
  P[idx] = s;
}

// biasx[n] = b_in[n] + sum_k conv_b_k @ Wx_blk  (constant across t)
__global__ void k_biasx(const float* __restrict__ b2, const float* __restrict__ b3,
                        const float* __restrict__ b4, const float* __restrict__ b5,
                        const float* __restrict__ Wx, const float* __restrict__ bin,
                        float* __restrict__ biasx)
{
  int n = blockIdx.x * blockDim.x + threadIdx.x;
  if (n >= NG) return;
  const float* bk[4] = {b2, b3, b4, b5};
  float s = bin[n];
  for (int kk = 0; kk < 4; ++kk) {
    const float* wx = Wx + (size_t)(kk * CHID) * NG + n;
    for (int c = 0; c < CHID; ++c) s += bk[kk][c] * wx[(size_t)c * NG];
  }
  biasx[n] = s;
}

// Whb[n][k] = (bf16) Wh[k][n]   (transpose so B-operand reads are contiguous)
__global__ void k_whb(const float* __restrict__ Wh, __bf16* __restrict__ Whb)
{
  int idx = blockIdx.x * blockDim.x + threadIdx.x;
  if (idx >= HIDN * NG) return;
  int k = idx & (HIDN - 1);
  int n = idx >> 8;
  Whb[(size_t)n * HIDN + k] = (__bf16)Wh[(size_t)k * NG + n];
}

// ---------------------------------------------------------------------------
extern "C" void kernel_launch(void* const* d_in, const int* in_sizes, int n_in,
                              void* d_out, int out_size, void* d_ws, size_t ws_size,
                              hipStream_t stream)
{
  const int*   X    = (const int*)  d_in[0];
  const float* emb  = (const float*)d_in[1];
  const float* w2   = (const float*)d_in[2];
  const float* b2   = (const float*)d_in[3];
  const float* w3   = (const float*)d_in[4];
  const float* b3   = (const float*)d_in[5];
  const float* w4   = (const float*)d_in[6];
  const float* b4   = (const float*)d_in[7];
  const float* w5   = (const float*)d_in[8];
  const float* b5   = (const float*)d_in[9];
  const float* Wx   = (const float*)d_in[10];
  const float* Wh   = (const float*)d_in[11];
  const float* bin  = (const float*)d_in[12];
  const float* brec = (const float*)d_in[13];
  float* out = (float*)d_out;

  char* ws = (char*)d_ws;
  const size_t offE  = 0;                       // 5*64*768*4   =   983,040 B
  const size_t offP  = offE + 983040;           // 5*129*768*4  = 1,981,440 B
  const size_t offBx = offP + 1981440;          // 768*4        =     3,072 B
  const size_t offWh = offBx + 3072;            // 768*256*2    =   393,216 B
  float*  E     = (float*)(ws + offE);
  float*  P     = (float*)(ws + offP);
  float*  biasx = (float*)(ws + offBx);
  __bf16* Whb   = (__bf16*)(ws + offWh);

  k_eff  <<<(5 * EMBD * NG + 255) / 256, 256, 0, stream>>>(w2, w3, w4, w5, Wx, E);
  k_ptab <<<(5 * NCHP * NG + 255) / 256, 256, 0, stream>>>(emb, E, P);
  k_biasx<<<(NG + 255) / 256,            256, 0, stream>>>(b2, b3, b4, b5, Wx, bin, biasx);
  k_whb  <<<(HIDN * NG + 255) / 256,     256, 0, stream>>>(Wh, Whb);
  k_gru  <<<8, 512, 0, stream>>>(X, P, biasx, Whb, brec, out);
}
